// GlobalRouting_69913477644423
// MI455X (gfx1250) — matrix operand order repaired and checked by reading, not compiled, not run on hardware
//
#include <hip/hip_runtime.h>
#include <hip/hip_bf16.h>
#include <math.h>

typedef __attribute__((ext_vector_type(16))) _Float16 v16h;
typedef __attribute__((ext_vector_type(8)))  _Float16 v8h;
typedef __attribute__((ext_vector_type(8)))  float    v8f;

#define KKA   288
#define NSITE 196
#define NTHR  288   // 9 waves: 18 m-tiles -> exactly 2 per wave (uniform trip count)

__device__ __forceinline__ float gelu_exact(float x) {
    return 0.5f * x * (1.0f + erff(x * 0.70710678118654752f));
}

// ---------------------------------------------------------------------------
// c_d[delta] = (1/288)[Re w0 + (-1)^delta Re w144 + 2 sum_{f=1..143}(Re wf cos - Im wf sin)]
// cw layout: [145][16][2]
__global__ void build_circ(const float* __restrict__ cw, float* __restrict__ c) {
    int idx = blockIdx.x * blockDim.x + threadIdx.x;
    if (idx >= 16 * 288) return;
    int d = idx / 288, dl = idx % 288;
    const float w0 = 6.283185307179586f / 288.0f;
    float s = cw[(0 * 16 + d) * 2 + 0];
    float nyq = cw[(144 * 16 + d) * 2 + 0];
    s += (dl & 1) ? -nyq : nyq;
    for (int f = 1; f < 144; ++f) {
        int ph = (f * dl) % 288;
        float ang = w0 * (float)ph;
        float re = cw[(f * 16 + d) * 2 + 0];
        float im = cw[(f * 16 + d) * 2 + 1];
        s += 2.0f * (re * cosf(ang) - im * sinf(ang));
    }
    c[idx] = s * (1.0f / 288.0f);
}

// G[d][m][k] = c_d[(m-k) mod 288] + (m==k)   (f16)
__global__ void build_G(const float* __restrict__ c, _Float16* __restrict__ G) {
    int idx = blockIdx.x * blockDim.x + threadIdx.x;
    if (idx >= 16 * 288 * 288) return;
    int d = idx / (288 * 288);
    int r = idx % (288 * 288);
    int m = r / 288, k = r % 288;
    int dl = m - k; if (dl < 0) dl += 288;
    float v = c[d * 288 + dl] + (m == k ? 1.0f : 0.0f);
    G[idx] = (_Float16)v;
}

__global__ void cvt_w(const float* __restrict__ mw, const float* __restrict__ c2w,
                      _Float16* __restrict__ mwh, _Float16* __restrict__ cwh) {
    int idx = blockIdx.x * blockDim.x + threadIdx.x;
    if (idx < 4096) mwh[idx] = (_Float16)mw[idx];
    if (idx < 256)  cwh[idx] = (_Float16)c2w[idx];
}

// ---------------------------------------------------------------------------
// One block (9 waves, 288 threads) per site i = b*196 + l.
__global__ __launch_bounds__(NTHR)
void routing_main(const float* __restrict__ a_in,
                  const float* __restrict__ pose,
                  const float* __restrict__ mpose_b,
                  const float* __restrict__ cpose2_b,
                  const float* __restrict__ ln_gamma,
                  const float* __restrict__ ln_beta,
                  const float* __restrict__ sp_w,
                  const float* __restrict__ sp_b,
                  const _Float16* __restrict__ G,
                  const _Float16* __restrict__ mwh,
                  const _Float16* __restrict__ cwh,
                  float* __restrict__ out)
{
    __shared__ _Float16 s_pu[KKA * 16];   // pu  [m][c]        f16
    __shared__ float    s_lg[KKA * 16];   // logit -> r -> coeff
    __shared__ float    s_nm[KKA * 16];   // normalized gate
    __shared__ _Float16 s_Pd[16 * KKA];   // P_d [b_][m]       f16
    __shared__ float    s_au[KKA];
    __shared__ float    s_po[256];
    __shared__ float    s_arsum[16];
    __shared__ float    s_ausum;

    const int tid  = threadIdx.x;
    const int site = blockIdx.x;
    const int b    = site / NSITE;
    const int l    = site % NSITE;
    const int y    = l / 14, x = l % 14;

    // ---- unfold pose -> pu (m = kk*32 + a, c), unfold a -> au
    for (int e = tid; e < KKA * 16; e += NTHR) {
        int m = e >> 4, cc = e & 15;
        int kk = m >> 5, ach = m & 31;
        int yy = y + (kk / 3) - 1, xx = x + (kk % 3) - 1;
        float v = 0.0f;
        if (yy >= 0 && yy < 14 && xx >= 0 && xx < 14)
            v = pose[((b * 512 + ach * 16 + cc) * 14 + yy) * 14 + xx];
        s_pu[e] = (_Float16)v;
    }
    {
        int m = tid;   // NTHR == KKA: exactly one element each
        int kk = m >> 5, ach = m & 31;
        int yy = y + (kk / 3) - 1, xx = x + (kk % 3) - 1;
        float v = 0.0f;
        if (yy >= 0 && yy < 14 && xx >= 0 && xx < 14)
            v = a_in[((b * 32 + ach) * 14 + yy) * 14 + xx];
        s_au[m] = v;
    }
    if (tid < 256) s_po[tid] = 0.0f;
    __syncthreads();

    const int wave = tid >> 5, lane = tid & 31;
    const int g = lane >> 4, ln = lane & 15;

    // ---- logit = pu @ cpose2_w^T + b  via WMMA (K=16 zero-padded to 32)
    {
        v16h vb;
        for (int i = 0; i < 16; ++i)
            vb[i] = (i < 8) ? cwh[ln * 16 + 8 * g + i] : (_Float16)0.0f;
        float bias = cpose2_b[ln];
        for (int t = 0; t < 2; ++t) {          // uniform trip count: no EXEC masking
            int mt = wave * 2 + t;
            int row = mt * 16 + ln;
            v8h lo = *(const v8h*)(s_pu + row * 16 + 8 * g);
            v16h va;
            for (int i = 0; i < 8; ++i) { va[i] = lo[i]; va[8 + i] = (_Float16)0.0f; }
            v8f acc = {0.f,0.f,0.f,0.f,0.f,0.f,0.f,0.f};
            acc = __builtin_amdgcn_wmma_f32_16x16x32_f16(false, va, false, vb,
                                                         (short)0, acc, false, false);
            for (int r = 0; r < 8; ++r)
                s_lg[(mt * 16 + 8 * g + r) * 16 + ln] = acc[r] + bias;
        }
    }
    __syncthreads();

    // ---- group norm over A (32) per (kk, b_)
    if (tid < 144) {
        int kk = tid >> 4, b_ = tid & 15;
        float s = 0.f, s2 = 0.f;
        for (int ach = 0; ach < 32; ++ach) {
            float v = s_lg[(kk * 32 + ach) * 16 + b_];
            s += v; s2 += v * v;
        }
        float mu  = s * (1.0f / 32.0f);
        float var = s2 * (1.0f / 32.0f) - mu * mu;
        float inv = rsqrtf(var + 1e-5f);
        for (int ach = 0; ach < 32; ++ach) {
            int m = kk * 32 + ach;
            float v = s_lg[m * 16 + b_];
            s_nm[m * 16 + b_] = (v - mu) * inv * ln_gamma[ach] + ln_beta[ach];
        }
    }
    __syncthreads();

    // ---- 9x9 kernel mixing + gelu; final logit = 2*logit + gelu_out
    for (int e = tid; e < KKA * 16; e += NTHR) {
        int m = e >> 4, b_ = e & 15;
        int o = m >> 5, ach = m & 31;
        float s = sp_b[o];
        for (int i = 0; i < 9; ++i)
            s += sp_w[o * 9 + i] * s_nm[(i * 32 + ach) * 16 + b_];
        s_lg[e] = 2.0f * s_lg[e] + gelu_exact(s);
    }
    __syncthreads();

    // ---- softmax over b_ per m (one m per thread)
    {
        int m = tid;
        float mx = -1e30f;
        for (int b_ = 0; b_ < 16; ++b_) mx = fmaxf(mx, s_lg[m * 16 + b_]);
        float e16[16], sum = 0.f;
        for (int b_ = 0; b_ < 16; ++b_) { e16[b_] = expf(s_lg[m * 16 + b_] - mx); sum += e16[b_]; }
        float inv = 1.0f / sum;
        for (int b_ = 0; b_ < 16; ++b_) s_lg[m * 16 + b_] = e16[b_] * inv;
    }
    __syncthreads();

    // ---- ar_sum[b_], au_sum
    if (tid < 16) {
        float s = 0.f;
        for (int m = 0; m < KKA; ++m) s += s_au[m] * s_lg[m * 16 + tid];
        s_arsum[tid] = s;
    } else if (tid == 16) {
        float s = 0.f;
        for (int m = 0; m < KKA; ++m) s += s_au[m];
        s_ausum = s;
    }
    __syncthreads();

    // ---- a_out and coeff = au*r / ar_sum
    if (tid < 16)
        out[(b * 16 + tid) * NSITE + l] = s_arsum[tid] / s_ausum;
    for (int e = tid; e < KKA * 16; e += NTHR) {
        int m = e >> 4, b_ = e & 15;
        s_lg[e] = s_au[m] * s_lg[e] / s_arsum[b_];
    }
    __syncthreads();

    // ---- per-d: P_d = gelu(pu @ mpose_w^T) ; Y_d = G_d @ P_d ; po += coeff * Y_d
    for (int d = 0; d < 16; ++d) {
        {   // build P_d via WMMA (K=16 padded)
            v16h vb;
            for (int i = 0; i < 16; ++i)
                vb[i] = (i < 8) ? mwh[(ln * 16 + d) * 16 + 8 * g + i] : (_Float16)0.0f;
            float bias = mpose_b[ln * 16 + d];
            for (int t = 0; t < 2; ++t) {
                int mt = wave * 2 + t;
                int row = mt * 16 + ln;
                v8h lo = *(const v8h*)(s_pu + row * 16 + 8 * g);
                v16h va;
                for (int i = 0; i < 8; ++i) { va[i] = lo[i]; va[8 + i] = (_Float16)0.0f; }
                v8f acc = {0.f,0.f,0.f,0.f,0.f,0.f,0.f,0.f};
                acc = __builtin_amdgcn_wmma_f32_16x16x32_f16(false, va, false, vb,
                                                             (short)0, acc, false, false);
                v8h st;
                for (int r = 0; r < 8; ++r) st[r] = (_Float16)gelu_exact(acc[r] + bias);
                *(v8h*)(s_Pd + ln * KKA + mt * 16 + 8 * g) = st;
            }
        }
        __syncthreads();
        {   // circulant filter GEMM, fold routing reduction out of accumulators
            const _Float16* Gd = G + (size_t)d * 288 * 288;
            for (int t = 0; t < 2; ++t) {
                int mt = wave * 2 + t;
                int row = mt * 16 + ln;
                const _Float16* Grow = Gd + (size_t)row * 288;
                v8f acc = {0.f,0.f,0.f,0.f,0.f,0.f,0.f,0.f};
                for (int ks = 0; ks < 9; ++ks) {
                    int k0 = ks * 32;
                    v8h alo = *(const v8h*)(Grow + k0 + 16 * g);
                    v8h ahi = *(const v8h*)(Grow + k0 + 16 * g + 8);
                    v16h va;
                    for (int i = 0; i < 8; ++i) { va[i] = alo[i]; va[8 + i] = ahi[i]; }
                    v16h vbb = *(const v16h*)(s_Pd + ln * KKA + k0 + 16 * g);
                    acc = __builtin_amdgcn_wmma_f32_16x16x32_f16(false, va, false, vbb,
                                                                 (short)0, acc, false, false);
                }
                float part = 0.f;
                for (int r = 0; r < 8; ++r)
                    part += acc[r] * s_lg[(mt * 16 + 8 * g + r) * 16 + ln];
                atomicAdd(&s_po[ln * 16 + d], part);
            }
        }
        __syncthreads();
    }

    // ---- write po: out channel j = b_*16+d, NCHW
    if (tid < 256)
        out[12544 + ((size_t)b * 256 + tid) * NSITE + l] = s_po[tid];
}

// ---------------------------------------------------------------------------
extern "C" void kernel_launch(void* const* d_in, const int* in_sizes, int n_in,
                              void* d_out, int out_size, void* d_ws, size_t ws_size,
                              hipStream_t stream)
{
    (void)in_sizes; (void)n_in; (void)out_size; (void)ws_size;
    const float* a_in = (const float*)d_in[0];
    const float* pose = (const float*)d_in[1];
    const float* mpw  = (const float*)d_in[2];
    const float* mpb  = (const float*)d_in[3];
    const float* c2w  = (const float*)d_in[4];
    const float* c2b  = (const float*)d_in[5];
    const float* cwgt = (const float*)d_in[6];
    const float* gam  = (const float*)d_in[7];
    const float* bet  = (const float*)d_in[8];
    const float* spw  = (const float*)d_in[9];
    const float* spb  = (const float*)d_in[10];

    char* ws = (char*)d_ws;
    float*    c_tab = (float*)(ws);               // 16*288 f32   = 18432 B
    _Float16* mwh   = (_Float16*)(ws + 32768);    // 256*16 f16   =  8192 B
    _Float16* cwh   = (_Float16*)(ws + 49152);    // 16*16  f16   =   512 B
    _Float16* G     = (_Float16*)(ws + 65536);    // 16*288*288   = 2.65 MB

    build_circ<<<(16 * 288 + 255) / 256, 256, 0, stream>>>(cwgt, c_tab);
    cvt_w<<<(4096 + 255) / 256, 256, 0, stream>>>(mpw, c2w, mwh, cwh);
    build_G<<<(16 * 288 * 288 + 255) / 256, 256, 0, stream>>>(c_tab, G);
    routing_main<<<784, NTHR, 0, stream>>>(a_in, pose, mpb, c2b, gam, bet, spw, spb,
                                           G, mwh, cwh, (float*)d_out);
}
